// QuantLinear_47708496724666
// MI455X (gfx1250) — matrix operand order repaired and checked by reading, not compiled, not run on hardware
//
#include <hip/hip_runtime.h>

// ---------------- problem dims ----------------
constexpr int M = 8192;   // B*S = 4*2048
constexpr int N = 4096;   // DOUT
constexpr int K = 4096;   // DIN

// ---------------- gemm tiling -----------------
constexpr int BM = 256;             // workgroup tile M
constexpr int BN = 128;             // workgroup tile N
constexpr int BK = 64;              // K per LDS stage
constexpr int LDSK = BK + 8;        // padded LDS row stride (halves); 144B rows, 16B aligned
constexpr int NSTAGE = K / BK;      // 64 K-stages
// 8 waves: 4 (M) x 2 (N), each wave owns a 64x64 tile = 4x4 fragments of 16x16

typedef __attribute__((ext_vector_type(16))) __bf16 v16bf;
typedef __attribute__((ext_vector_type(8)))  __bf16 v8bf;
typedef __attribute__((ext_vector_type(8)))  float  v8f;
typedef int v4i_ __attribute__((vector_size(16)));   // matches async-lds builtin param type

#if __has_builtin(__builtin_amdgcn_global_load_async_to_lds_b128)
#define USE_ASYNC 1
#define ASG __attribute__((address_space(1)))
#define ASL __attribute__((address_space(3)))
#else
#define USE_ASYNC 0
#endif

__device__ __forceinline__ void wait_async0() {
#if __has_builtin(__builtin_amdgcn_s_wait_asynccnt)
  __builtin_amdgcn_s_wait_asynccnt(0);
#else
  asm volatile("s_wait_asynccnt 0" ::: "memory");
#endif
}

// ---------------- kernel 1: x fp32 -> bf16 ----------------
__global__ __launch_bounds__(256)
void k_cvt_x(const float* __restrict__ x, __bf16* __restrict__ xb) {
  size_t idx = ((size_t)blockIdx.x * 256 + threadIdx.x) * 8;
  const float4* p = (const float4*)(x + idx);
  float4 u = p[0], v = p[1];
  v8bf r;
  r[0] = (__bf16)u.x; r[1] = (__bf16)u.y; r[2] = (__bf16)u.z; r[3] = (__bf16)u.w;
  r[4] = (__bf16)v.x; r[5] = (__bf16)v.y; r[6] = (__bf16)v.z; r[7] = (__bf16)v.w;
  *(v8bf*)(xb + idx) = r;
}

// ---------------- kernel 2: W int8 * scale -> bf16 ----------------
__global__ __launch_bounds__(256)
void k_dequant(const signed char* __restrict__ qw, const float* __restrict__ scale,
               __bf16* __restrict__ wb) {
  size_t idx = (size_t)blockIdx.x * 256 + threadIdx.x;   // 8-elem chunk id
  int row = (int)(idx >> 9);                             // K/8 = 512 chunks per row
  float s = scale[row];
  int2 raw = *(const int2*)(qw + idx * 8);
  v8bf r;
#pragma unroll
  for (int i = 0; i < 8; ++i) {
    int w = (i < 4) ? raw.x : raw.y;
    int q = (int)(signed char)((w >> ((i & 3) * 8)) & 0xFF);
    r[i] = (__bf16)(s * (float)q);
  }
  *(v8bf*)(wb + idx * 8) = r;
}

// ---------------- kernel 3: bf16 WMMA GEMM ----------------
__global__ __launch_bounds__(256)
void k_gemm(const __bf16* __restrict__ Xb, const __bf16* __restrict__ Wb,
            const float* __restrict__ bias, __bf16* __restrict__ out) {
  __shared__ __align__(16) __bf16 sA[2][BM * LDSK];   // 2 * 256*72*2B = 72 KB
  __shared__ __align__(16) __bf16 sB[2][BN * LDSK];   // 2 * 128*72*2B = 36 KB

  const int t    = threadIdx.x;
  const int lane = t & 31;
  const int wave = t >> 5;
  const int wm   = wave >> 1;          // 0..3  (M direction, 64 rows each)
  const int wn   = wave & 1;           // 0..1  (N direction, 64 cols each)
  const int lr   = lane & 15;          // row/col within a 16-wide fragment
  const int hi   = lane >> 4;          // lane half

  const int m0 = blockIdx.y * BM;
  const int n0 = blockIdx.x * BN;

  // global->LDS chunk mapping (16B chunks of 8 halves):
  // A: 256 rows x 8 chunks = 2048 chunks -> 8/thread; B: 128 rows x 8 = 1024 -> 4/thread
  int arow[8], acol[8];
#pragma unroll
  for (int c = 0; c < 8; ++c) {
    int lin = t + 256 * c;
    arow[c] = lin >> 3;
    acol[c] = (lin & 7) * 8;
  }

  const __bf16* gA = Xb + (size_t)m0 * K;
  const __bf16* gB = Wb + (size_t)n0 * K;

  v8f acc[4][4] = {};

#if !USE_ASYNC
  v8bf stA[8], stB[4];
#endif

#if USE_ASYNC
  // prologue: issue stage 0
#pragma unroll
  for (int c = 0; c < 8; ++c)
    __builtin_amdgcn_global_load_async_to_lds_b128(
        (ASG v4i_*)(gA + (size_t)arow[c] * K + acol[c]),
        (ASL v4i_*)(&sA[0][arow[c] * LDSK + acol[c]]), 0, 0);
#pragma unroll
  for (int c = 0; c < 4; ++c)
    __builtin_amdgcn_global_load_async_to_lds_b128(
        (ASG v4i_*)(gB + (size_t)arow[c] * K + acol[c]),
        (ASL v4i_*)(&sB[0][arow[c] * LDSK + acol[c]]), 0, 0);
#else
#pragma unroll
  for (int c = 0; c < 8; ++c)
    stA[c] = *(const v8bf*)(gA + (size_t)arow[c] * K + acol[c]);
#pragma unroll
  for (int c = 0; c < 4; ++c)
    stB[c] = *(const v8bf*)(gB + (size_t)arow[c] * K + acol[c]);
#pragma unroll
  for (int c = 0; c < 8; ++c)
    *(v8bf*)(&sA[0][arow[c] * LDSK + acol[c]]) = stA[c];
#pragma unroll
  for (int c = 0; c < 4; ++c)
    *(v8bf*)(&sB[0][arow[c] * LDSK + acol[c]]) = stB[c];
#endif

  for (int kt = 0; kt < NSTAGE; ++kt) {
    const int buf = kt & 1;

#if USE_ASYNC
    wait_async0();            // this wave's async LDS writes complete
    __syncthreads();          // all waves' writes visible; prior reads of buf^1 done
    if (kt + 1 < NSTAGE) {
      const int k1 = (kt + 1) * BK;
#pragma unroll
      for (int c = 0; c < 8; ++c)
        __builtin_amdgcn_global_load_async_to_lds_b128(
            (ASG v4i_*)(gA + (size_t)arow[c] * K + k1 + acol[c]),
            (ASL v4i_*)(&sA[buf ^ 1][arow[c] * LDSK + acol[c]]), 0, 0);
#pragma unroll
      for (int c = 0; c < 4; ++c)
        __builtin_amdgcn_global_load_async_to_lds_b128(
            (ASG v4i_*)(gB + (size_t)arow[c] * K + k1 + acol[c]),
            (ASL v4i_*)(&sB[buf ^ 1][arow[c] * LDSK + acol[c]]), 0, 0);
    }
#else
    if (kt + 1 < NSTAGE) {
      const int k1 = (kt + 1) * BK;
#pragma unroll
      for (int c = 0; c < 8; ++c)
        stA[c] = *(const v8bf*)(gA + (size_t)arow[c] * K + k1 + acol[c]);
#pragma unroll
      for (int c = 0; c < 4; ++c)
        stB[c] = *(const v8bf*)(gB + (size_t)arow[c] * K + k1 + acol[c]);
    }
    __syncthreads();
#endif

    // ---- compute on buf: 2 k-steps of 32, 16 WMMAs each (16 ds_load_b128 each) ----
    const __bf16* pA = &sA[buf][0];
    const __bf16* pB = &sB[buf][0];
#pragma unroll
    for (int ks = 0; ks < 2; ++ks) {
      const int kk = ks * 32;
      v16bf bfr[4];
#pragma unroll
      for (int j = 0; j < 4; ++j) {
        // B frag 32x16: lane = N column; lanes 0-15 hold K 0-15, lanes 16-31 hold K 16-31
        const __bf16* p = pB + (wn * 64 + j * 16 + lr) * LDSK + kk + hi * 16;
        v8bf lo = *(const v8bf*)p;
        v8bf h2 = *(const v8bf*)(p + 8);
        bfr[j] = __builtin_shufflevector(lo, h2, 0,1,2,3,4,5,6,7,8,9,10,11,12,13,14,15);
      }
#pragma unroll
      for (int i = 0; i < 4; ++i) {
        // A frag 16x32: lanes 0-15 hold K 0-15, lanes 16-31 hold K 16-31
        const __bf16* p = pA + (wm * 64 + i * 16 + lr) * LDSK + kk + hi * 16;
        v8bf lo = *(const v8bf*)p;
        v8bf h2 = *(const v8bf*)(p + 8);
        v16bf af = __builtin_shufflevector(lo, h2, 0,1,2,3,4,5,6,7,8,9,10,11,12,13,14,15);
#pragma unroll
        for (int j = 0; j < 4; ++j)
          acc[i][j] = __builtin_amdgcn_wmma_f32_16x16x32_bf16(
              false, af, false, bfr[j], (short)0, acc[i][j], false, false);
      }
    }

#if !USE_ASYNC
    __syncthreads();
    if (kt + 1 < NSTAGE) {
#pragma unroll
      for (int c = 0; c < 8; ++c)
        *(v8bf*)(&sA[buf ^ 1][arow[c] * LDSK + acol[c]]) = stA[c];
#pragma unroll
      for (int c = 0; c < 4; ++c)
        *(v8bf*)(&sB[buf ^ 1][arow[c] * LDSK + acol[c]]) = stB[c];
    }
#endif
  }

  // ---- epilogue: C layout is VGPR r -> M = r + 8*hi, N = lane%16 ----
  const size_t m_base = (size_t)m0 + wm * 64;
  const int n_base = n0 + wn * 64;
#pragma unroll
  for (int j = 0; j < 4; ++j) {
    const int col = n_base + j * 16 + lr;
    const float bv = bias[col];
#pragma unroll
    for (int i = 0; i < 4; ++i) {
#pragma unroll
      for (int r = 0; r < 8; ++r) {
        size_t row = m_base + i * 16 + hi * 8 + r;
        out[row * N + col] = (__bf16)(acc[i][j][r] + bv);
      }
    }
  }
}

// ---------------- launcher ----------------
extern "C" void kernel_launch(void* const* d_in, const int* in_sizes, int n_in,
                              void* d_out, int out_size, void* d_ws, size_t ws_size,
                              hipStream_t stream) {
  const float*       x     = (const float*)d_in[0];
  const signed char* qw    = (const signed char*)d_in[1];
  const float*       scale = (const float*)d_in[2];
  const float*       bias  = (const float*)d_in[3];
  __bf16*            out   = (__bf16*)d_out;

  // workspace: X_bf16 (64 MB) then W_bf16 (32 MB)
  __bf16* Xb = (__bf16*)d_ws;
  __bf16* Wb = Xb + (size_t)M * K;

  k_cvt_x  <<<(int)(((size_t)M * K) / 8 / 256), 256, 0, stream>>>(x, Xb);
  k_dequant<<<(int)(((size_t)N * K) / 8 / 256), 256, 0, stream>>>(qw, scale, Wb);
  k_gemm   <<<dim3(N / BN, M / BM), 256, 0, stream>>>(Xb, Wb, bias, out);
}
